// Attention_5781025980924
// MI455X (gfx1250) — compile-verified, matrix operand order checked by reading, not run on hardware
//
#include <hip/hip_runtime.h>
#include <math.h>

// ---------------------------------------------------------------------------
// Types for CDNA5 WMMA (wave32): bf16 operands, f32 accumulate.
// ---------------------------------------------------------------------------
typedef __bf16 bf16_t;
typedef __attribute__((ext_vector_type(8)))  __bf16 v8bf;
typedef __attribute__((ext_vector_type(16))) __bf16 v16bf;
typedef __attribute__((ext_vector_type(8)))  float  v8f;

typedef __attribute__((ext_vector_type(4))) unsigned int u32x4;
typedef __attribute__((ext_vector_type(4))) int          i32x4;
typedef __attribute__((ext_vector_type(8))) int          i32x8;

#define S_LEN  4096
#define HID    1024
#define NHEADS 16
#define HDIM   64

#if defined(__HIP_DEVICE_COMPILE__) && __has_builtin(__builtin_amdgcn_tensor_load_to_lds)
#define USE_TDM 1
#else
#define USE_TDM 0
#endif

static __device__ __forceinline__ v16bf frag16(const bf16_t* lo, const bf16_t* hi) {
  v8bf a = *(const v8bf*)lo;
  v8bf b = *(const v8bf*)hi;
  return __builtin_shufflevector(a, b, 0,1,2,3,4,5,6,7,8,9,10,11,12,13,14,15);
}

static __device__ __forceinline__ v8f wmma_bf16(v16bf a, v16bf b, v8f c) {
  // D = A(16x32) * B(32x16) + C(16x16), f32 accum
  return __builtin_amdgcn_wmma_f32_16x16x32_bf16(false, a, false, b, (short)0, c, false, false);
}

static __device__ __forceinline__ unsigned lds_off(const void* p) {
  // Generic pointers to LDS carry the LDS byte offset in addr[31:0].
  return (unsigned)(unsigned long long)(uintptr_t)p;
}

// ---------------------------------------------------------------------------
// Tensor Data Mover: 2D tile load Global -> LDS.
// D# per ISA 8.3/8.4: data_size=2B, tile (tile_d0 x tile_d1), row stride in
// elements; rows packed contiguously at lds_addr.  Wave-level op (EXEC
// ignored), tracked with TENSORcnt.
// ---------------------------------------------------------------------------
static __device__ __forceinline__ void tdm_load_2d(
    const void* gptr, unsigned lds_byte_addr,
    unsigned tile_d0, unsigned tile_d1,
    unsigned long long row_stride_elems)
{
#if USE_TDM
  unsigned long long ga = (unsigned long long)(uintptr_t)gptr;
  unsigned td0 = (unsigned)row_stride_elems;  // full rows valid
  unsigned td1 = 0x40000000u;                 // effectively no dim1 clipping

  u32x4 g0;
  g0[0] = 1u;                                            // count=1, user D#
  g0[1] = lds_byte_addr;                                 // lds_addr
  g0[2] = (unsigned)(ga & 0xffffffffull);                // global_addr[31:0]
  g0[3] = (unsigned)((ga >> 32) & 0x01ffffffull) | (2u << 30);  // [56:32]|type=2

  i32x8 g1;
  g1[0] = (int)(1u << 16);                               // data_size=1 -> 2B
  g1[1] = (int)((td0 & 0xffffu) << 16);                  // tensor_dim0[15:0]
  g1[2] = (int)(((td0 >> 16) & 0xffffu) | ((td1 & 0xffffu) << 16));
  g1[3] = (int)(((td1 >> 16) & 0xffffu) | (tile_d0 << 16));   // tile_dim0
  g1[4] = (int)(tile_d1 & 0xffffu);                      // tile_dim1 (dim2=0)
  g1[5] = (int)(row_stride_elems & 0xffffffffull);       // dim0_stride[31:0]
  g1[6] = (int)((row_stride_elems >> 32) & 0xffffull);   // [47:32] (dim1_stride=0)
  g1[7] = 0;

  i32x4 z4 = {0, 0, 0, 0};
#if __clang_major__ >= 23
  i32x8 z8 = {0, 0, 0, 0, 0, 0, 0, 0};
  __builtin_amdgcn_tensor_load_to_lds(g0, g1, z4, z4, z8, 0);
#else
  __builtin_amdgcn_tensor_load_to_lds(g0, g1, z4, z4, 0);
#endif
#else
  (void)gptr; (void)lds_byte_addr; (void)tile_d0; (void)tile_d1; (void)row_stride_elems;
#endif
}

static __device__ __forceinline__ void wait_tensor_le2() {
#if USE_TDM
#if __has_builtin(__builtin_amdgcn_s_wait_tensorcnt)
  __builtin_amdgcn_s_wait_tensorcnt(2);
#else
  asm volatile("s_wait_tensorcnt 0x2" ::: "memory");
#endif
#endif
}
static __device__ __forceinline__ void wait_tensor_le0() {
#if USE_TDM
#if __has_builtin(__builtin_amdgcn_s_wait_tensorcnt)
  __builtin_amdgcn_s_wait_tensorcnt(0);
#else
  asm volatile("s_wait_tensorcnt 0x0" ::: "memory");
#endif
#endif
}

// ---------------------------------------------------------------------------
// fp32 -> bf16 conversion (4 elements / thread; all sizes are multiples of 4)
// ---------------------------------------------------------------------------
__global__ void cvt_f32_to_bf16(const float* __restrict__ in,
                                bf16_t* __restrict__ out, int n) {
  int i = (blockIdx.x * blockDim.x + threadIdx.x) * 4;
  if (i < n) {
    float4 v = *(const float4*)(in + i);
    out[i + 0] = (bf16_t)v.x;
    out[i + 1] = (bf16_t)v.y;
    out[i + 2] = (bf16_t)v.z;
    out[i + 3] = (bf16_t)v.w;
  }
}

// ---------------------------------------------------------------------------
// Tiled WMMA GEMM:  C[M][N] = A[M][K] * B^T  where B is stored [N][K]
// (k-contiguous for both operands).  256 threads = 8 waves; each wave owns a
// 32(M) x 64(N) C tile -> 2x4 accumulators of 16x16.  TDM double-buffered.
// mode 0: bf16 C row-major (with scale)   mode 1: bf16 C stored C^T [N][M]
// mode 2: f32  C row-major
// ---------------------------------------------------------------------------
#define BM 128
#define BN 128
#define BK 32

static __device__ __forceinline__ void gemm_step(
    const bf16_t* __restrict__ As, const bf16_t* __restrict__ Bs,
    v8f (&acc)[2][4], int wmi, int wni, int l16, int klo)
{
  v16bf af[2], bfr[4];
  for (int mi = 0; mi < 2; ++mi) {
    int r = wmi * 32 + mi * 16 + l16;
    af[mi] = frag16(&As[r * BK + klo], &As[r * BK + klo + 16]);
  }
  for (int ni = 0; ni < 4; ++ni) {
    int r = wni * 64 + ni * 16 + l16;
    bfr[ni] = frag16(&Bs[r * BK + klo], &Bs[r * BK + klo + 16]);
  }
  for (int mi = 0; mi < 2; ++mi)
    for (int ni = 0; ni < 4; ++ni)
      acc[mi][ni] = wmma_bf16(af[mi], bfr[ni], acc[mi][ni]);
}

__global__ __launch_bounds__(256) void gemm_bf16_wmma(
    const bf16_t* __restrict__ A,
    const bf16_t* __restrict__ B,
    void* __restrict__ Cout,
    int M, int N, int K, int mode, float scale)
{
  __shared__ bf16_t As[2][BM * BK];
  __shared__ bf16_t Bs[2][BN * BK];

  const int tid  = threadIdx.x;
  const int lane = tid & 31;
  const int wv   = tid >> 5;       // wave 0..7
  const int wmi  = wv >> 1;        // 0..3 : 32-row strip
  const int wni  = wv & 1;         // 0..1 : 64-col strip
  const int m0   = blockIdx.y * BM;
  const int n0   = blockIdx.x * BN;
  const int half = (lane >= 16);
  const int l16  = lane & 15;
  const int klo  = half ? 8 : 0;   // ISA A/B bf16 fragment K-base per lane half

  v8f acc[2][4];
  for (int mi = 0; mi < 2; ++mi)
    for (int ni = 0; ni < 4; ++ni)
      for (int r = 0; r < 8; ++r) acc[mi][ni][r] = 0.0f;

  int cur = 0;
#if USE_TDM
  if (wv == 0) {   // prologue: DMA first K-tile pair
    tdm_load_2d(&A[(size_t)m0 * K], lds_off(&As[0][0]), BK, BM, (unsigned long long)K);
    tdm_load_2d(&B[(size_t)n0 * K], lds_off(&Bs[0][0]), BK, BM, (unsigned long long)K);
  }
#endif

  for (int k0 = 0; k0 < K; k0 += BK) {
#if USE_TDM
    if (wv == 0) {
      if (k0 + BK < K) {   // DMA next tile pair into the other buffer
        tdm_load_2d(&A[(size_t)m0 * K + k0 + BK], lds_off(&As[cur ^ 1][0]),
                    BK, BM, (unsigned long long)K);
        tdm_load_2d(&B[(size_t)n0 * K + k0 + BK], lds_off(&Bs[cur ^ 1][0]),
                    BK, BM, (unsigned long long)K);
        wait_tensor_le2();   // in-order: current pair complete
      } else {
        wait_tensor_le0();
      }
    }
#else
    // Cooperative tile load: 512 chunks of 8 bf16 (16 B) each, 2 per thread.
    for (int i = 0; i < 2; ++i) {
      int c   = tid * 2 + i;        // 0..511
      int row = c >> 2;             // 0..127
      int kc  = (c & 3) << 3;       // 0,8,16,24
      *(v8bf*)&As[cur][row * BK + kc] =
          *(const v8bf*)&A[(size_t)(m0 + row) * K + k0 + kc];
      *(v8bf*)&Bs[cur][row * BK + kc] =
          *(const v8bf*)&B[(size_t)(n0 + row) * K + k0 + kc];
    }
    if (k0 + BK < K) {
      __builtin_prefetch(&A[(size_t)(m0 + (tid >> 1)) * K + k0 + BK], 0, 1);
      __builtin_prefetch(&B[(size_t)(n0 + (tid >> 1)) * K + k0 + BK], 0, 1);
    }
#endif
    __syncthreads();                       // tile ready for all waves
    gemm_step(As[cur], Bs[cur], acc, wmi, wni, l16, klo);
    __syncthreads();                       // all done reading before reuse
    cur ^= 1;
  }

  // C fragment: VGPR r -> row (r | r+8 per lane half), col = l16.
  for (int mi = 0; mi < 2; ++mi)
    for (int ni = 0; ni < 4; ++ni)
      for (int r = 0; r < 8; ++r) {
        int m = m0 + wmi * 32 + mi * 16 + (half ? r + 8 : r);
        int n = n0 + wni * 64 + ni * 16 + l16;
        float v = acc[mi][ni][r] * scale;
        if (mode == 0)      ((bf16_t*)Cout)[(size_t)m * N + n] = (bf16_t)v;
        else if (mode == 1) ((bf16_t*)Cout)[(size_t)n * M + m] = (bf16_t)v;
        else                ((float*)Cout)[(size_t)m * N + n]  = v;
      }
}

// ---------------------------------------------------------------------------
// Flash attention (causal).  Block = (head h, 128-query tile qb), 256 threads.
// Wave w owns query rows [qb*128 + w*16, +16).  KV chunks of 128, TDM
// double-buffered.  Q pre-scaled by 1/sqrt(HDIM).  Vt is [NHEADS][HDIM][S].
// ---------------------------------------------------------------------------
__global__ __launch_bounds__(256) void flash_attn_wmma(
    const bf16_t* __restrict__ Q,   // [S][HID]  (scaled)
    const bf16_t* __restrict__ Km,  // [S][HID]
    const bf16_t* __restrict__ Vt,  // [NHEADS][HDIM][S]
    bf16_t* __restrict__ O)         // [S][HID]
{
  __shared__ bf16_t Qs[128 * 64];            // 16 KB
  __shared__ bf16_t Ks[2][128 * 64];         // 32 KB
  __shared__ bf16_t Vs[2][64 * 128];         // 32 KB (V^T chunk: [d][kv])
  __shared__ bf16_t Ps[8 * 16 * 128];        // 32 KB (per-wave P tiles)

  const int qb   = blockIdx.x;
  const int h    = blockIdx.y;
  const int tid  = threadIdx.x;
  const int lane = tid & 31;
  const int wv   = tid >> 5;
  const int half = (lane >= 16);
  const int l16  = lane & 15;
  const int klo  = half ? 8 : 0;
  const int qrow0 = qb * 128 + wv * 16;

#if USE_TDM
  if (wv == 0) {   // prologue: DMA KV chunk 0 (overlaps Q load below)
    tdm_load_2d(&Km[(size_t)(0) * HID + h * HDIM], lds_off(&Ks[0][0]),
                HDIM, 128, (unsigned long long)HID);
    tdm_load_2d(&Vt[(size_t)h * HDIM * S_LEN], lds_off(&Vs[0][0]),
                128, HDIM, (unsigned long long)S_LEN);
  }
#endif

  // Load Q tile (128 x 64): 1024 chunks of 8 bf16, 4 per thread.
  for (int i = 0; i < 4; ++i) {
    int c = tid * 4 + i, row = c >> 3, kc = (c & 7) << 3;
    *(v8bf*)&Qs[row * 64 + kc] =
        *(const v8bf*)&Q[(size_t)(qb * 128 + row) * HID + h * HDIM + kc];
  }

  float mrow[8], lrow[8];
  v8f oacc[4];
  for (int r = 0; r < 8; ++r) { mrow[r] = -3.0e38f; lrow[r] = 0.0f; }
  for (int dt = 0; dt < 4; ++dt)
    for (int r = 0; r < 8; ++r) oacc[dt][r] = 0.0f;

  int cur = 0;
  for (int kc0 = 0; kc0 <= qb; ++kc0) {
#if USE_TDM
    if (wv == 0) {
      if (kc0 < qb) {   // DMA next KV chunk into the other buffer
        tdm_load_2d(&Km[(size_t)((kc0 + 1) * 128) * HID + h * HDIM],
                    lds_off(&Ks[cur ^ 1][0]), HDIM, 128, (unsigned long long)HID);
        tdm_load_2d(&Vt[(size_t)h * HDIM * S_LEN + (kc0 + 1) * 128],
                    lds_off(&Vs[cur ^ 1][0]), 128, HDIM, (unsigned long long)S_LEN);
        wait_tensor_le2();   // in-order: current chunk complete
      } else {
        wait_tensor_le0();
      }
    }
#else
    for (int i = 0; i < 4; ++i) {   // K chunk (128 x 64)
      int c = tid * 4 + i, row = c >> 3, kk = (c & 7) << 3;
      *(v8bf*)&Ks[cur][row * 64 + kk] =
          *(const v8bf*)&Km[(size_t)(kc0 * 128 + row) * HID + h * HDIM + kk];
    }
    for (int i = 0; i < 4; ++i) {   // V^T chunk (64 x 128)
      int c = tid * 4 + i, row = c >> 4, cc = (c & 15) << 3;
      *(v8bf*)&Vs[cur][row * 128 + cc] =
          *(const v8bf*)&Vt[((size_t)h * HDIM + row) * S_LEN + kc0 * 128 + cc];
    }
#endif
    __syncthreads();

    // S = Q * K^T  (16 WMMAs / wave)
    v8f sacc[8];
    for (int nt = 0; nt < 8; ++nt)
      for (int r = 0; r < 8; ++r) sacc[nt][r] = 0.0f;
    {
      int qr = wv * 16 + l16;
      v16bf aq0 = frag16(&Qs[qr * 64 + klo],      &Qs[qr * 64 + klo + 16]);
      v16bf aq1 = frag16(&Qs[qr * 64 + 32 + klo], &Qs[qr * 64 + 32 + klo + 16]);
      for (int nt = 0; nt < 8; ++nt) {
        int n = nt * 16 + l16;
        v16bf b0 = frag16(&Ks[cur][n * 64 + klo],      &Ks[cur][n * 64 + klo + 16]);
        v16bf b1 = frag16(&Ks[cur][n * 64 + 32 + klo], &Ks[cur][n * 64 + 32 + klo + 16]);
        sacc[nt] = wmma_bf16(aq0, b0, sacc[nt]);
        sacc[nt] = wmma_bf16(aq1, b1, sacc[nt]);
      }
    }

    // Causal mask + online softmax.  Row of VGPR r lives in one 16-lane half.
    for (int r = 0; r < 8; ++r) {
      int mglob = qrow0 + (half ? r + 8 : r);
      float mx = -3.0e38f;
      for (int nt = 0; nt < 8; ++nt) {
        int nglob = kc0 * 128 + nt * 16 + l16;
        float s = sacc[nt][r];
        s = (nglob <= mglob) ? s : -3.0e38f;
        sacc[nt][r] = s;
        mx = fmaxf(mx, s);
      }
      mx = fmaxf(mx, __shfl_xor(mx, 1));
      mx = fmaxf(mx, __shfl_xor(mx, 2));
      mx = fmaxf(mx, __shfl_xor(mx, 4));
      mx = fmaxf(mx, __shfl_xor(mx, 8));

      float nm   = fmaxf(mrow[r], mx);
      float corr = __expf(mrow[r] - nm);
      mrow[r] = nm;

      float rs = 0.0f;
      int prow = (half ? r + 8 : r);
      for (int nt = 0; nt < 8; ++nt) {
        float p = __expf(sacc[nt][r] - nm);
        rs += p;
        Ps[wv * 2048 + prow * 128 + nt * 16 + l16] = (bf16_t)p;
      }
      rs += __shfl_xor(rs, 1);
      rs += __shfl_xor(rs, 2);
      rs += __shfl_xor(rs, 4);
      rs += __shfl_xor(rs, 8);
      lrow[r] = lrow[r] * corr + rs;
      for (int dt = 0; dt < 4; ++dt) oacc[dt][r] *= corr;
    }
    __syncthreads();   // P visible / ordered before PV reads

    // O += P * V  (16 WMMAs / wave): A = P[16x128], B = Vs (d-major,kv-contig)
    for (int ks = 0; ks < 4; ++ks) {
      const bf16_t* pb = &Ps[wv * 2048 + l16 * 128 + ks * 32];
      v16bf ap = frag16(pb + klo, pb + klo + 16);
      for (int dt = 0; dt < 4; ++dt) {
        const bf16_t* vb = &Vs[cur][(dt * 16 + l16) * 128 + ks * 32];
        v16bf bv = frag16(vb + klo, vb + klo + 16);
        oacc[dt] = wmma_bf16(ap, bv, oacc[dt]);
      }
    }
    __syncthreads();   // done reading cur before next DMA overwrites buffers
    cur ^= 1;
  }

  // Normalize and store bf16 attention output.
  for (int dt = 0; dt < 4; ++dt)
    for (int r = 0; r < 8; ++r) {
      int m = qrow0 + (half ? r + 8 : r);
      int d = dt * 16 + l16;
      float v = oacc[dt][r] / lrow[r];
      O[(size_t)m * HID + h * HDIM + d] = (bf16_t)v;
    }
}

// ---------------------------------------------------------------------------
// Launch pipeline
// ---------------------------------------------------------------------------
extern "C" void kernel_launch(void* const* d_in, const int* in_sizes, int n_in,
                              void* d_out, int out_size, void* d_ws, size_t ws_size,
                              hipStream_t stream) {
  const float* x  = (const float*)d_in[0];
  const float* wq = (const float*)d_in[1];
  const float* wk = (const float*)d_in[2];
  const float* wv = (const float*)d_in[3];
  const float* wo = (const float*)d_in[4];
  float* out = (float*)d_out;

  const size_t SH = (size_t)S_LEN * HID;   // 4M elems
  const size_t WW = (size_t)HID * HID;     // 1M elems

  bf16_t* ws  = (bf16_t*)d_ws;
  size_t off  = 0;
  bf16_t* xb  = ws + off; off += SH;
  bf16_t* wqb = ws + off; off += WW;
  bf16_t* wkb = ws + off; off += WW;
  bf16_t* wvb = ws + off; off += WW;
  bf16_t* wob = ws + off; off += WW;
  bf16_t* Qb  = ws + off; off += SH;
  bf16_t* Kb  = ws + off; off += SH;
  bf16_t* Vtb = ws + off; off += SH;       // [NHEADS][HDIM][S]
  bf16_t* Ob  = ws + off; off += SH;

  // 1) fp32 -> bf16 conversions
  cvt_f32_to_bf16<<<(int)(SH / 4 / 256), 256, 0, stream>>>(x,  xb,  (int)SH);
  cvt_f32_to_bf16<<<(int)(WW / 4 / 256), 256, 0, stream>>>(wq, wqb, (int)WW);
  cvt_f32_to_bf16<<<(int)(WW / 4 / 256), 256, 0, stream>>>(wk, wkb, (int)WW);
  cvt_f32_to_bf16<<<(int)(WW / 4 / 256), 256, 0, stream>>>(wv, wvb, (int)WW);
  cvt_f32_to_bf16<<<(int)(WW / 4 / 256), 256, 0, stream>>>(wo, wob, (int)WW);

  // 2) projections (C = x * W^T; W stored [out][in] is already N-major/K-contig)
  dim3 ggrid(HID / BN, S_LEN / BM);
  const float scale = 0.125f;   // 1/sqrt(64)
  gemm_bf16_wmma<<<ggrid, 256, 0, stream>>>(xb, wqb, Qb,  S_LEN, HID, HID, 0, scale);
  gemm_bf16_wmma<<<ggrid, 256, 0, stream>>>(xb, wkb, Kb,  S_LEN, HID, HID, 0, 1.0f);
  gemm_bf16_wmma<<<ggrid, 256, 0, stream>>>(xb, wvb, Vtb, S_LEN, HID, HID, 1, 1.0f);

  // 3) causal flash attention
  flash_attn_wmma<<<dim3(S_LEN / 128, NHEADS), 256, 0, stream>>>(Qb, Kb, Vtb, Ob);

  // 4) output projection -> fp32 d_out
  gemm_bf16_wmma<<<ggrid, 256, 0, stream>>>(Ob, wob, out, S_LEN, HID, HID, 2, 1.0f);
}
